// MambaBlock2D_68315749810364
// MI455X (gfx1250) — compile-verified, matrix operand order checked by reading, not run on hardware
//
#include <hip/hip_runtime.h>
#include <math.h>

// ---------------- problem constants ----------------
#define DIMC   64
#define DI     128
#define DSN    16
#define HW     9216
#define NBATCH 2
#define MTOK   (NBATCH * HW)     // 18432 tokens
#define CHUNK  64
#define NCHUNK (HW / CHUNK)      // 144 chunks per batch

typedef __attribute__((ext_vector_type(16))) _Float16 v16h;
typedef __attribute__((ext_vector_type(8)))  float    v8f;
typedef unsigned uv4 __attribute__((ext_vector_type(4)));
typedef int      iv4 __attribute__((ext_vector_type(4)));
typedef int      iv8 __attribute__((ext_vector_type(8)));

union H2U { unsigned u; _Float16 h[2]; };

#if defined(__HIP_DEVICE_COMPILE__) && __has_builtin(__builtin_amdgcn_tensor_load_to_lds)
#define USE_TDM 1
#else
#define USE_TDM 0
#endif

// Load a 16x32 f16 A-fragment from an LDS row-major tile (ldk halfs per row).
// ISA 7.12.2 layout: lane = half*16 + m ; VGPR v holds K = (v/4)*16 + half*8 + (v%4)*2 (+0,+1)
__device__ __forceinline__ v16h wmma_load_a(const _Float16* base, int ldk, int lane) {
    int m = lane & 15, half = lane >> 4;
    const _Float16* row = base + m * ldk;
    v16h a;
#pragma unroll
    for (int v = 0; v < 8; ++v) {
        int k = ((v >> 2) << 4) + (half << 3) + ((v & 3) << 1);
        H2U p; p.u = *(const unsigned*)(row + k);   // K,K+1 adjacent -> one b32 LDS read
        a[2 * v]     = p.h[0];
        a[2 * v + 1] = p.h[1];
    }
    return a;
}

// Stage n contiguous f16 elements (n < 65536, multiple of 8) from global into LDS.
// TDM path: 1-D descriptor — tensor n x 1, tile n x 1, data_size = 2B, stride0 = n.
// Issued by one wave; per-wave TENSORcnt waited before the workgroup barrier.
__device__ __forceinline__ void stage_tile(_Float16* lds, const _Float16* src, int n) {
#if USE_TDM
    if (threadIdx.x < 32) {
        if (threadIdx.x == 0) {
            unsigned ldsOff = (unsigned)(size_t)(void*)lds;
            unsigned long long ga = (unsigned long long)(size_t)src;
            uv4 g0 = { 1u,                                   // count=1, user descriptor
                       ldsOff,                               // D#.lds_addr
                       (unsigned)ga,                         // global_addr[31:0]
                       (unsigned)((ga >> 32) & 0x01FFFFFFull) | 0x80000000u }; // addr[56:32] | type=2
            iv8 g1 = { 0x00010000,                           // workgroup_mask=0, data_size=1 (2B)
                       (int)((unsigned)(n & 0xFFFF) << 16),  // tensor_dim0[15:0]
                       0x00010000 | (n >> 16),               // tensor_dim0[31:16] | tensor_dim1=1
                       (int)((unsigned)(n & 0xFFFF) << 16),  // tile_dim0
                       0,                                    // tile_dim1=0, tile_dim2=0
                       n,                                    // tensor_dim0_stride[31:0]
                       0, 0 };
            iv4 gz = { 0, 0, 0, 0 };
#if defined(__clang_major__) && (__clang_major__ >= 23)
            iv8 gz8 = { 0, 0, 0, 0, 0, 0, 0, 0 };
            __builtin_amdgcn_tensor_load_to_lds(g0, g1, gz, gz, gz8, 0);
#else
            __builtin_amdgcn_tensor_load_to_lds(g0, g1, gz, gz, 0);
#endif
        }
        __builtin_amdgcn_s_wait_tensorcnt(0);  // s_wait_tensorcnt 0 (issuing wave)
    }
#else
    const float4* s4 = (const float4*)src;
    float4* d4 = (float4*)lds;
    for (int i = threadIdx.x; i < n / 8; i += blockDim.x) d4[i] = s4[i];
#endif
    __syncthreads();
}

__device__ __forceinline__ float siluf(float v)  { return v / (1.f + __expf(-v)); }
__device__ __forceinline__ float clip10(float v) { return fminf(10.f, fmaxf(-10.f, v)); }

// ---------------- prep kernels ----------------
__global__ void k_zero4(float* stats) { if (threadIdx.x < 4) stats[threadIdx.x] = 0.f; }

__global__ void k_stats(const float* __restrict__ x, float* stats) {
    __shared__ float red[1024];
    float s0 = 0, q0 = 0, s1 = 0, q1 = 0;
    const int NE = DIMC * HW;
    int tid = blockIdx.x * blockDim.x + threadIdx.x;
    int stride = gridDim.x * blockDim.x;
    for (int i = tid; i < NBATCH * NE; i += stride) {
        float v = x[i];
        if (i < NE) { s0 += v; q0 += v * v; } else { s1 += v; q1 += v * v; }
    }
    red[threadIdx.x] = s0; red[256 + threadIdx.x] = q0;
    red[512 + threadIdx.x] = s1; red[768 + threadIdx.x] = q1;
    __syncthreads();
    for (int o = 128; o > 0; o >>= 1) {
        if (threadIdx.x < o) {
            red[threadIdx.x] += red[threadIdx.x + o];
            red[256 + threadIdx.x] += red[256 + threadIdx.x + o];
            red[512 + threadIdx.x] += red[512 + threadIdx.x + o];
            red[768 + threadIdx.x] += red[768 + threadIdx.x + o];
        }
        __syncthreads();
    }
    if (threadIdx.x == 0) {
        atomicAdd(&stats[0], red[0]);   atomicAdd(&stats[1], red[256]);
        atomicAdd(&stats[2], red[512]); atomicAdd(&stats[3], red[768]);
    }
}

// Pack W (N x K, row major f32) into WMMA B-fragments: B[k][n] = W[n][k].
__global__ void k_pack_b(const float* __restrict__ W, _Float16* __restrict__ Bp,
                         int N, int K, int Ntiles, int Ktiles) {
    int idx = blockIdx.x * blockDim.x + threadIdx.x;
    int total = Ntiles * Ktiles * 512;
    if (idx >= total) return;
    int j = idx & 15, lane = (idx >> 4) & 31, tile = idx >> 9;
    int kt = tile % Ktiles, nt = tile / Ktiles;
    int half = lane >> 4, n = nt * 16 + (lane & 15);
    int v = j >> 1, pos = j & 1;
    int k = kt * 32 + ((v >> 2) << 4) + (half << 3) + ((v & 3) << 1) + pos;
    float val = (n < N && k < K) ? W[n * K + k] : 0.f;
    Bp[idx] = (_Float16)val;
}

__global__ void k_aneg(const float* __restrict__ A_log, float* __restrict__ Aneg) {
    int i = blockIdx.x * blockDim.x + threadIdx.x;
    if (i < DI * DSN) Aneg[i] = -__expf(A_log[i]);
}

// ---------------- norm + NCHW -> token-major transpose ----------------
__global__ void k_norm(const float* __restrict__ x, const float* __restrict__ g,
                       const float* __restrict__ be, const float* __restrict__ stats,
                       float* __restrict__ res_tok, _Float16* __restrict__ seqh) {
    __shared__ float tile[64][65];
    int blk = blockIdx.x;
    int b = blk / NCHUNK;
    int hw0 = (blk % NCHUNK) * 64;
    float N = (float)(DIMC * HW);
    float mu = stats[b * 2 + 0] / N;
    float var = stats[b * 2 + 1] / N - mu * mu;
    float rstd = rsqrtf(var + 1e-5f);
    for (int i = threadIdx.x; i < 64 * 64; i += blockDim.x) {
        int c = i >> 6, j = i & 63;
        float v = x[((size_t)(b * DIMC + c)) * HW + hw0 + j];
        v = clip10((v - mu) * rstd * g[c] + be[c]);
        tile[c][j] = v;
    }
    __syncthreads();
    for (int i = threadIdx.x; i < 64 * 64; i += blockDim.x) {
        int j = i >> 6, c = i & 63;
        float v = tile[c][j];
        size_t row = (size_t)(b * HW + hw0 + j);
        res_tok[row * 64 + c] = v;
        seqh[row * 64 + c] = (_Float16)v;
    }
}

// ---------------- GEMM1: xz = seq @ W_in^T + b_in  (M x 256, K=64) ----------------
__global__ void k_gemm_xz(const _Float16* __restrict__ Ah, const _Float16* __restrict__ Bp,
                          const float* __restrict__ bias,
                          float* __restrict__ xraw, float* __restrict__ zbuf) {
    __shared__ __align__(16) _Float16 As[64 * 64];
    int m0 = blockIdx.x * 64;
    stage_tile(As, Ah + (size_t)m0 * 64, 64 * 64);
    int lane = threadIdx.x & 31, wave = threadIdx.x >> 5;   // 8 waves
    const v8f vz = {};
    v8f acc[4][2];
#pragma unroll
    for (int a = 0; a < 4; ++a) { acc[a][0] = vz; acc[a][1] = vz; }
#pragma unroll
    for (int kt = 0; kt < 2; ++kt) {
        v16h af[4];
#pragma unroll
        for (int mt = 0; mt < 4; ++mt) af[mt] = wmma_load_a(As + mt * 16 * 64 + kt * 32, 64, lane);
#pragma unroll
        for (int jn = 0; jn < 2; ++jn) {
            int nt = wave * 2 + jn;
            v16h bf = *(const v16h*)(Bp + ((size_t)(nt * 2 + kt)) * 512 + lane * 16);
#pragma unroll
            for (int mt = 0; mt < 4; ++mt)
                acc[mt][jn] = __builtin_amdgcn_wmma_f32_16x16x32_f16(
                    false, af[mt], false, bf, (short)0, acc[mt][jn], false, false);
        }
    }
    int half = lane >> 4, nl = lane & 15;
#pragma unroll
    for (int jn = 0; jn < 2; ++jn) {
        int n = (wave * 2 + jn) * 16 + nl;
        float bv = bias[n];
#pragma unroll
        for (int mt = 0; mt < 4; ++mt) {
#pragma unroll
            for (int r = 0; r < 8; ++r) {
                int m = m0 + mt * 16 + half * 8 + r;
                float v = acc[mt][jn][r] + bv;
                if (n < DI) xraw[(size_t)m * DI + n] = v;
                else        zbuf[(size_t)m * DI + (n - DI)] = v;
            }
        }
    }
}

// ---------------- depthwise causal conv4 + SiLU ----------------
__global__ void k_conv(const float* __restrict__ xraw, const float* __restrict__ cw,
                       const float* __restrict__ cb,
                       float* __restrict__ xa32, _Float16* __restrict__ xah) {
    int idx = blockIdx.x * blockDim.x + threadIdx.x;
    if (idx >= MTOK * DI) return;
    int d = idx & 127, m = idx >> 7;
    int b = m / HW, l = m - b * HW;
    float acc = cb[d];
#pragma unroll
    for (int t = 0; t < 4; ++t) {
        int ls = l - 3 + t;
        if (ls >= 0) acc += cw[d * 4 + t] * xraw[(size_t)(b * HW + ls) * DI + d];
    }
    float s = siluf(acc);
    xa32[idx] = s;
    xah[idx] = (_Float16)s;
}

// ---------------- GEMM2: dbc = xa @ W_xproj^T  (M x 48pad, K=128) ----------------
__global__ void k_gemm_xproj(const _Float16* __restrict__ Ah, const _Float16* __restrict__ Bp,
                             float* __restrict__ dbc) {
    __shared__ __align__(16) _Float16 As[64 * 128];
    int m0 = blockIdx.x * 64;
    stage_tile(As, Ah + (size_t)m0 * 128, 64 * 128);
    int lane = threadIdx.x & 31, wave = threadIdx.x >> 5;   // 4 waves, one M-tile each
    const v8f vz = {};
    v8f acc[3] = { vz, vz, vz };
#pragma unroll
    for (int kt = 0; kt < 4; ++kt) {
        v16h af = wmma_load_a(As + wave * 16 * 128 + kt * 32, 128, lane);
#pragma unroll
        for (int nt = 0; nt < 3; ++nt) {
            v16h bf = *(const v16h*)(Bp + ((size_t)(nt * 4 + kt)) * 512 + lane * 16);
            acc[nt] = __builtin_amdgcn_wmma_f32_16x16x32_f16(
                false, af, false, bf, (short)0, acc[nt], false, false);
        }
    }
    int half = lane >> 4, nl = lane & 15;
#pragma unroll
    for (int nt = 0; nt < 3; ++nt) {
        int n = nt * 16 + nl;
#pragma unroll
        for (int r = 0; r < 8; ++r) {
            int m = m0 + wave * 16 + half * 8 + r;
            dbc[(size_t)m * 48 + n] = acc[nt][r];
        }
    }
}

// ---------------- dt = softplus(dtr @ W_dt^T + b_dt) ----------------
__global__ void k_dt(const float* __restrict__ dbc, const float* __restrict__ Wdt,
                     const float* __restrict__ bdt, float* __restrict__ dt) {
    int idx = blockIdx.x * blockDim.x + threadIdx.x;
    if (idx >= MTOK * DI) return;
    int d = idx & 127, m = idx >> 7;
    const float* r = dbc + (size_t)m * 48;
    float a = bdt[d] + r[0] * Wdt[d * 4] + r[1] * Wdt[d * 4 + 1]
                     + r[2] * Wdt[d * 4 + 2] + r[3] * Wdt[d * 4 + 3];
    dt[idx] = (a > 20.f) ? a : log1pf(__expf(a));
}

// ---------------- scan phase A: per-chunk (prod a, h_end) ----------------
__global__ void k_scanA(const float* __restrict__ dtb, const _Float16* __restrict__ xah,
                        const float* __restrict__ dbc, const float* __restrict__ Aneg,
                        float* __restrict__ chA, float* __restrict__ chH) {
    __shared__ float    dtS[CHUNK * DI];
    __shared__ _Float16 xaS[CHUNK * DI];
    __shared__ float    BmS[CHUNK * DSN];
    int blk = blockIdx.x;
    int b = blk / NCHUNK, ch = blk % NCHUNK;
    int mBase = b * HW + ch * CHUNK;
    for (int i = threadIdx.x; i < CHUNK * DI; i += blockDim.x) {
        dtS[i] = dtb[(size_t)mBase * DI + i];
        xaS[i] = xah[(size_t)mBase * DI + i];
    }
    for (int i = threadIdx.x; i < CHUNK * DSN; i += blockDim.x) {
        int l = i >> 4, s = i & 15;
        BmS[i] = dbc[(size_t)(mBase + l) * 48 + 4 + s];
    }
    __syncthreads();
    int t = threadIdx.x;                 // 256 threads: (d, 8-state half)
    int d = t >> 1, sb = (t & 1) * 8;
    float Ar[8], h[8], aP[8];
#pragma unroll
    for (int s = 0; s < 8; ++s) { Ar[s] = Aneg[d * DSN + sb + s]; h[s] = 0.f; aP[s] = 1.f; }
    for (int l = 0; l < CHUNK; ++l) {
        float dtv = dtS[l * DI + d];
        float xv = (float)xaS[l * DI + d];
        float dx = dtv * xv;
#pragma unroll
        for (int s = 0; s < 8; ++s) {
            float da = __expf(dtv * Ar[s]);
            h[s] = da * h[s] + dx * BmS[l * DSN + sb + s];
            aP[s] *= da;
        }
    }
    size_t base = (size_t)blk * 2048 + d * DSN + sb;
#pragma unroll
    for (int s = 0; s < 8; ++s) { chA[base + s] = aP[s]; chH[base + s] = h[s]; }
}

// ---------------- scan phase B: sequential inter-chunk scan ----------------
__global__ void k_scanB(const float* __restrict__ chA, const float* __restrict__ chH,
                        float* __restrict__ hInit) {
    int ch = blockIdx.x * blockDim.x + threadIdx.x;
    if (ch >= NBATCH * 2048) return;
    int b = ch >> 11, c2 = ch & 2047;
    float h = 0.f;
    for (int c = 0; c < NCHUNK; ++c) {
        size_t i = (size_t)(b * NCHUNK + c) * 2048 + c2;
        if (c + 1 < NCHUNK) {
            __builtin_prefetch(&chA[i + 2048], 0, 1);   // global_prefetch_b8
            __builtin_prefetch(&chH[i + 2048], 0, 1);
        }
        hInit[i] = h;
        h = chA[i] * h + chH[i];
    }
}

// ---------------- scan phase C: replay + C-proj + D skip + SiLU(z) gate ----------------
__global__ void k_scanC(const float* __restrict__ dtb, const _Float16* __restrict__ xah,
                        const float* __restrict__ dbc, const float* __restrict__ Aneg,
                        const float* __restrict__ hInit, const float* __restrict__ Dp,
                        const float* __restrict__ zbuf, _Float16* __restrict__ ygh) {
    __shared__ float    dtS[CHUNK * DI];
    __shared__ _Float16 xaS[CHUNK * DI];
    __shared__ float    BmS[CHUNK * DSN];
    __shared__ float    CmS[CHUNK * DSN];
    int blk = blockIdx.x;
    int b = blk / NCHUNK, ch = blk % NCHUNK;
    int mBase = b * HW + ch * CHUNK;
    for (int i = threadIdx.x; i < CHUNK * DI; i += blockDim.x) {
        dtS[i] = dtb[(size_t)mBase * DI + i];
        xaS[i] = xah[(size_t)mBase * DI + i];
    }
    for (int i = threadIdx.x; i < CHUNK * DSN; i += blockDim.x) {
        int l = i >> 4, s = i & 15;
        BmS[i] = dbc[(size_t)(mBase + l) * 48 + 4 + s];
        CmS[i] = dbc[(size_t)(mBase + l) * 48 + 20 + s];
    }
    __syncthreads();
    int t = threadIdx.x;
    int d = t >> 1, sb = (t & 1) * 8;
    float Ar[8], h[8];
    size_t hbase = (size_t)blk * 2048 + d * DSN + sb;
#pragma unroll
    for (int s = 0; s < 8; ++s) { Ar[s] = Aneg[d * DSN + sb + s]; h[s] = hInit[hbase + s]; }
    float dpv = Dp[d];
    for (int l = 0; l < CHUNK; ++l) {
        float dtv = dtS[l * DI + d];
        float xv = (float)xaS[l * DI + d];
        float dx = dtv * xv;
        float acc = 0.f;
#pragma unroll
        for (int s = 0; s < 8; ++s) {
            float da = __expf(dtv * Ar[s]);
            h[s] = da * h[s] + dx * BmS[l * DSN + sb + s];
            acc += h[s] * CmS[l * DSN + sb + s];
        }
        acc += __shfl_xor(acc, 1, 32);          // combine the two 8-state halves
        if ((t & 1) == 0) {
            int m = mBase + l;
            float y = acc + xv * dpv;
            float zv = zbuf[(size_t)m * DI + d];
            ygh[(size_t)m * DI + d] = (_Float16)(y * siluf(zv));
        }
    }
}

// ---------------- chained GEMMs: (yg @ W_out^T + b_out) @ proj_w^T + epilogue ----------------
__global__ void k_gemm_chain(const _Float16* __restrict__ Ah, const _Float16* __restrict__ BpO,
                             const float* __restrict__ bout, const _Float16* __restrict__ BpP,
                             const float* __restrict__ pbias,
                             const float* __restrict__ bng, const float* __restrict__ bnb,
                             const float* __restrict__ bnm, const float* __restrict__ bnv,
                             const float* __restrict__ res_tok, float* __restrict__ out) {
    __shared__ __align__(16) _Float16 As[64 * 128];
    __shared__ __align__(16) _Float16 Ts[64 * 64];
    int m0 = blockIdx.x * 64;
    stage_tile(As, Ah + (size_t)m0 * 128, 64 * 128);
    int lane = threadIdx.x & 31, wave = threadIdx.x >> 5;   // 8 waves
    int mt = wave >> 1, np = wave & 1;
    int half = lane >> 4, nl = lane & 15;
    const v8f vz = {};
    // stage 1: T = yg @ W_out^T + b_out   (64 x 64, K=128)
    v8f acc[2] = { vz, vz };
#pragma unroll
    for (int kt = 0; kt < 4; ++kt) {
        v16h af = wmma_load_a(As + mt * 16 * 128 + kt * 32, 128, lane);
#pragma unroll
        for (int j = 0; j < 2; ++j) {
            int nt = np * 2 + j;
            v16h bf = *(const v16h*)(BpO + ((size_t)(nt * 4 + kt)) * 512 + lane * 16);
            acc[j] = __builtin_amdgcn_wmma_f32_16x16x32_f16(
                false, af, false, bf, (short)0, acc[j], false, false);
        }
    }
#pragma unroll
    for (int j = 0; j < 2; ++j) {
        int n = (np * 2 + j) * 16 + nl;
        float bv = bout[n];
#pragma unroll
        for (int r = 0; r < 8; ++r)
            Ts[(mt * 16 + half * 8 + r) * 64 + n] = (_Float16)(acc[j][r] + bv);
    }
    __syncthreads();
    // stage 2: P = T @ proj_w^T   (64 x 64, K=64)
    v8f pacc[2] = { vz, vz };
#pragma unroll
    for (int kt = 0; kt < 2; ++kt) {
        v16h af = wmma_load_a(Ts + mt * 16 * 64 + kt * 32, 64, lane);
#pragma unroll
        for (int j = 0; j < 2; ++j) {
            int nt = np * 2 + j;
            v16h bf = *(const v16h*)(BpP + ((size_t)(nt * 2 + kt)) * 512 + lane * 16);
            pacc[j] = __builtin_amdgcn_wmma_f32_16x16x32_f16(
                false, af, false, bf, (short)0, pacc[j], false, false);
        }
    }
    // epilogue: BN -> exact GELU -> residual -> clip -> NCHW
#pragma unroll
    for (int j = 0; j < 2; ++j) {
        int n = (np * 2 + j) * 16 + nl;
        float mean = bnm[n], rstd = rsqrtf(bnv[n] + 1e-5f);
        float ga = bng[n], bb = bnb[n], pb = pbias[n];
#pragma unroll
        for (int r = 0; r < 8; ++r) {
            int m = m0 + mt * 16 + half * 8 + r;
            float p = pacc[j][r] + pb;
            p = (p - mean) * rstd * ga + bb;
            float ge = 0.5f * p * (1.f + erff(p * 0.70710678f));
            float v = clip10(res_tok[(size_t)m * 64 + n] + ge);
            int bt = m / HW, l = m - bt * HW;
            out[((size_t)(bt * DIMC + n)) * HW + l] = v;
        }
    }
}

// ---------------- host launcher ----------------
extern "C" void kernel_launch(void* const* d_in, const int* in_sizes, int n_in,
                              void* d_out, int out_size, void* d_ws, size_t ws_size,
                              hipStream_t stream) {
    const float* x       = (const float*)d_in[0];
    const float* gn_g    = (const float*)d_in[1];
    const float* gn_b    = (const float*)d_in[2];
    const float* W_in    = (const float*)d_in[3];
    const float* b_in    = (const float*)d_in[4];
    const float* conv_w  = (const float*)d_in[5];
    const float* conv_b  = (const float*)d_in[6];
    const float* W_xproj = (const float*)d_in[7];
    const float* W_dt    = (const float*)d_in[8];
    const float* b_dt    = (const float*)d_in[9];
    const float* A_log   = (const float*)d_in[10];
    const float* Dp      = (const float*)d_in[11];
    const float* W_out   = (const float*)d_in[12];
    const float* b_out   = (const float*)d_in[13];
    const float* proj_w  = (const float*)d_in[14];
    const float* proj_b  = (const float*)d_in[15];
    const float* bn_g    = (const float*)d_in[16];
    const float* bn_b    = (const float*)d_in[17];
    const float* bn_m    = (const float*)d_in[18];
    const float* bn_v    = (const float*)d_in[19];
    float* out = (float*)d_out;

    unsigned char* ws = (unsigned char*)d_ws;
    size_t off = 0;
    auto alloc = [&](size_t bytes) -> void* {
        off = (off + 255) & ~(size_t)255;
        void* p = ws + off;
        off += bytes;
        return p;
    };
    float*     stats   = (float*)    alloc(4 * sizeof(float));
    _Float16*  seqh    = (_Float16*) alloc((size_t)MTOK * 64 * 2);
    float*     res_tok = (float*)    alloc((size_t)MTOK * 64 * 4);
    float*     xraw    = (float*)    alloc((size_t)MTOK * DI * 4);
    float*     zbuf    = (float*)    alloc((size_t)MTOK * DI * 4);
    float*     xa32    = (float*)    alloc((size_t)MTOK * DI * 4);
    _Float16*  xah     = (_Float16*) alloc((size_t)MTOK * DI * 2);
    float*     dbc     = (float*)    alloc((size_t)MTOK * 48 * 4);
    float*     dtb     = (float*)    alloc((size_t)MTOK * DI * 4);
    float*     chA     = (float*)    alloc((size_t)NBATCH * NCHUNK * 2048 * 4);
    float*     chH     = (float*)    alloc((size_t)NBATCH * NCHUNK * 2048 * 4);
    float*     hInit   = (float*)    alloc((size_t)NBATCH * NCHUNK * 2048 * 4);
    _Float16*  ygh     = (_Float16*) alloc((size_t)MTOK * DI * 2);
    _Float16*  BpIn    = (_Float16*) alloc((size_t)16 * 2 * 512 * 2);
    _Float16*  BpXp    = (_Float16*) alloc((size_t)3 * 4 * 512 * 2);
    _Float16*  BpOut   = (_Float16*) alloc((size_t)4 * 4 * 512 * 2);
    _Float16*  BpPr    = (_Float16*) alloc((size_t)4 * 2 * 512 * 2);
    float*     Aneg    = (float*)    alloc((size_t)DI * DSN * 4);
    if (off > ws_size) return;   // workspace too small: nothing safe to do

    // prep
    k_zero4<<<1, 64, 0, stream>>>(stats);
    k_stats<<<256, 256, 0, stream>>>(x, stats);
    k_pack_b<<<(16 * 2 * 512 + 255) / 256, 256, 0, stream>>>(W_in,    BpIn, 256, 64, 16, 2);
    k_pack_b<<<(3 * 4 * 512 + 255) / 256, 256, 0, stream>>>(W_xproj, BpXp, 36, 128, 3, 4);
    k_pack_b<<<(4 * 4 * 512 + 255) / 256, 256, 0, stream>>>(W_out,   BpOut, 64, 128, 4, 4);
    k_pack_b<<<(4 * 2 * 512 + 255) / 256, 256, 0, stream>>>(proj_w,  BpPr, 64, 64, 4, 2);
    k_aneg<<<(DI * DSN + 255) / 256, 256, 0, stream>>>(A_log, Aneg);

    // pipeline
    k_norm<<<NBATCH * NCHUNK, 256, 0, stream>>>(x, gn_g, gn_b, stats, res_tok, seqh);
    k_gemm_xz<<<MTOK / 64, 256, 0, stream>>>(seqh, BpIn, b_in, xraw, zbuf);
    k_conv<<<(MTOK * DI + 255) / 256, 256, 0, stream>>>(xraw, conv_w, conv_b, xa32, xah);
    k_gemm_xproj<<<MTOK / 64, 128, 0, stream>>>(xah, BpXp, dbc);
    k_dt<<<(MTOK * DI + 255) / 256, 256, 0, stream>>>(dbc, W_dt, b_dt, dtb);
    k_scanA<<<NBATCH * NCHUNK, 256, 0, stream>>>(dtb, xah, dbc, Aneg, chA, chH);
    k_scanB<<<(NBATCH * 2048 + 255) / 256, 256, 0, stream>>>(chA, chH, hInit);
    k_scanC<<<NBATCH * NCHUNK, 256, 0, stream>>>(dtb, xah, dbc, Aneg, hInit, Dp, zbuf, ygh);
    k_gemm_chain<<<MTOK / 64, 256, 0, stream>>>(ygh, BpOut, b_out, BpPr, proj_b,
                                                bn_g, bn_b, bn_m, bn_v, res_tok, out);
}
